// VGGQv2_83992380440633
// MI455X (gfx1250) — 8-bit matrix operand order repaired and checked by reading, compiled, not run on hardware
//
#include <hip/hip_runtime.h>
#include <hip/hip_bf16.h>
#include <stdint.h>

typedef __attribute__((ext_vector_type(8))) int v8i;
typedef __attribute__((ext_vector_type(4))) uint8_t u8x4;
typedef __attribute__((ext_vector_type(4))) unsigned int u32x4;

#define TRY_ASYNC 1

#define PING_OFF   ((size_t)0)
#define PONG_OFF   ((size_t)67108864)
#define WF_OFF     ((size_t)134217728)
#define SB_OFF     ((size_t)(134217728 + 4571136 + 256))

__global__ void scalebias_kernel(const float* __restrict__ g,
                                 const float* __restrict__ b,
                                 const float* __restrict__ m,
                                 const float* __restrict__ v,
                                 const float* __restrict__ wa,
                                 const float* __restrict__ aaPrev,
                                 float* __restrict__ S, float* __restrict__ B,
                                 int cout) {
  int oc = blockIdx.x * blockDim.x + threadIdx.x;
  if (oc >= cout) return;
  float s  = g[oc] * rsqrtf(v[oc] + 1e-5f);
  float sw = wa ? wa[oc] : 1.0f;
  float ap = aaPrev ? aaPrev[0] : 1.0f;
  S[oc] = s * sw * ap;
  B[oc] = b[oc] - m[oc] * s;
}

__global__ void wquant_kernel(const float* __restrict__ w,
                              const float* __restrict__ wa,
                              int8_t* __restrict__ frag,
                              int CIN, int logcin, int total) {
  int idx = blockIdx.x * blockDim.x + threadIdx.x;
  if (idx >= total) return;
  int KT   = (CIN * 9) >> 6;
  int j    = idx & 31;
  int l    = (idx >> 5) & 31;
  int tile = idx >> 10;
  int kt = tile % KT, ot = tile / KT;
  int m = l & 15, half = l >> 4, vv = j >> 2, bb = j & 3;
  int K_local = ((j >> 4) << 5) + (half << 4) + (j & 15);
  int k   = (kt << 6) + K_local;
  int tap = k >> logcin;
  int ci  = k & (CIN - 1);
  int oc  = ot * 16 + m;
  float a = wa[oc];
  float q = rintf(fminf(fmaxf(w[((size_t)oc * CIN + ci) * 9 + tap] / a, -8.f), 7.f));
  frag[idx] = (int8_t)q;
}

__global__ __launch_bounds__(256) void conv0_kernel(const float* __restrict__ x,
                                                    const float* __restrict__ w,
                                                    const float* __restrict__ S,
                                                    const float* __restrict__ B,
                                                    const float* __restrict__ aa,
                                                    uint8_t* __restrict__ out) {
  __shared__ float wl[3456];
  for (int i = threadIdx.x; i < 3456; i += 256) {
    int tap = i >> 7, oc = i & 127;
    wl[i] = w[oc * 27 + tap];
  }
  __syncthreads();
  int idx = blockIdx.x * 256 + threadIdx.x;
  int oc  = idx & 127;
  int pix = (idx >> 7) & 1023;
  int n   = idx >> 17;
  int y = pix >> 5, xx = pix & 31;
  const float* xi = x + (size_t)n * 3 * 1024;
  float acc = 0.f;
  #pragma unroll
  for (int ci = 0; ci < 3; ++ci)
    #pragma unroll
    for (int ky = 0; ky < 3; ++ky) {
      int yy = y + ky - 1;
      if (yy < 0 || yy >= 32) continue;
      #pragma unroll
      for (int kx = 0; kx < 3; ++kx) {
        int xs = xx + kx - 1;
        if (xs < 0 || xs >= 32) continue;
        acc += xi[ci * 1024 + yy * 32 + xs] * wl[(ci * 9 + ky * 3 + kx) * 128 + oc];
      }
    }
  float inv = 1.0f / aa[0];
  float h = acc * S[oc] + B[oc];
  float q = fminf(fmaxf(rintf(h * inv), 0.f), 15.f);
  out[idx] = (uint8_t)q;
}

template <int CIN, int COUT, int H, int W, int OCW>
__global__ __launch_bounds__(256) void convq_kernel(const uint8_t* __restrict__ act,
                                                    const int8_t* __restrict__ wfrag,
                                                    const float* __restrict__ Sc,
                                                    const float* __restrict__ Bc,
                                                    const float* __restrict__ aa,
                                                    uint8_t* __restrict__ out) {
  constexpr int CB     = CIN / 64;
  constexpr int KT     = 9 * CB;
  constexpr int HW     = H * W;
  constexpr int LOGW   = (W == 32) ? 5 : ((W == 16) ? 4 : 3);
  constexpr int LOGCIN = (CIN == 128) ? 7 : ((CIN == 256) ? 8 : 9);
  __shared__ __attribute__((aligned(32))) int btile[2][4][256];

  const int tid  = threadIdx.x;
  const int wave = tid >> 5;
  const int lane = tid & 31;
  const int pix0 = blockIdx.x * 64;
  const int ocT0 = (blockIdx.y * 8 + wave) * OCW;

  const int img     = pix0 / HW;
  const int rembase = pix0 % HW;
  const uint8_t* actImg = act + (size_t)img * HW * CIN;

  const int ntG = tid >> 6;
  const int rG  = tid & 63;
  const int lG  = rG >> 1;
  const int qG  = rG & 1;
  const int nG  = lG & 15;
  const int hG  = lG >> 4;
  const int kloc = qG * 32 + hG * 16;

  const int rpG = rembase + ntG * 16 + nG;
  const int pyG = rpG >> LOGW;
  const int pxG = rpG & (W - 1);
  const int ldsSlot = ntG * 256 + lG * 8 + qG * 4;

  auto gather = [&](int kt, int buf) {
    int t  = kt / CB;
    int cb = kt % CB;
    int ky = t / 3, kx = t - ky * 3;
    int yy = pyG + ky - 1;
    int xs = pxG + kx - 1;
    bool ok = ((unsigned)yy < (unsigned)H) & ((unsigned)xs < (unsigned)W);
    int yc = min(max(yy, 0), H - 1);
    int xc = min(max(xs, 0), W - 1);
    unsigned int addr =
        ((unsigned)(((yc << LOGW) + xc)) << LOGCIN) + (unsigned)((cb << 6) + kloc);
#if TRY_ASYNC
    unsigned int ldsoff = (unsigned int)((buf * 1024 + ldsSlot) * 4);
    if (ok) {
      const void* gsrc = (const void*)(actImg + addr);
      asm volatile("global_load_async_to_lds_b128 %0, %1, off"
                   :: "v"(ldsoff), "v"(gsrc) : "memory");
    } else {
      *(u32x4*)&btile[buf][ntG][lG * 8 + qG * 4] = (u32x4)(0u);
    }
#else
    u32x4 val = *(const u32x4*)(actImg + addr);
    val = ok ? val : (u32x4)(0u);
    *(u32x4*)&btile[buf][ntG][lG * 8 + qG * 4] = val;
#endif
  };

  v8i acc[OCW][4];
  #pragma unroll
  for (int ot = 0; ot < OCW; ++ot)
    #pragma unroll
    for (int nt = 0; nt < 4; ++nt) acc[ot][nt] = (v8i){0, 0, 0, 0, 0, 0, 0, 0};

  const int8_t* wbase[OCW];
  #pragma unroll
  for (int ot = 0; ot < OCW; ++ot)
    wbase[ot] = wfrag + (size_t)(ocT0 + ot) * KT * 1024 + lane * 32;

  gather(0, 0);
#if TRY_ASYNC
  asm volatile("s_wait_asynccnt 0x0" ::: "memory");
#endif
  __syncthreads();

  #pragma unroll 1
  for (int kt = 0; kt < KT; ++kt) {
    const int buf = kt & 1;
    if (kt + 1 < KT) {
      gather(kt + 1, buf ^ 1);
      #pragma unroll
      for (int ot = 0; ot < OCW; ++ot)
        __builtin_prefetch(wbase[ot] + (size_t)(kt + 1) * 1024, 0, 3);
    }
    v8i a[OCW];
    #pragma unroll
    for (int ot = 0; ot < OCW; ++ot)
      a[ot] = *(const v8i*)(wbase[ot] + (size_t)kt * 1024);
    v8i bf[4];
    #pragma unroll
    for (int nt = 0; nt < 4; ++nt)
      bf[nt] = *(const v8i*)&btile[buf][nt][lane * 8];
    #pragma unroll
    for (int nt = 0; nt < 4; ++nt)
      #pragma unroll
      for (int ot = 0; ot < OCW; ++ot)
        acc[ot][nt] = __builtin_amdgcn_wmma_i32_16x16x64_iu8(
             true, a[ot],  false, bf[nt], acc[ot][nt],
            false, false);
#if TRY_ASYNC
    asm volatile("s_wait_asynccnt 0x0" ::: "memory");
#endif
    __syncthreads();
  }

  const float inv = 1.0f / aa[0];
  const int n2 = lane & 15;
  const int hf = lane >> 4;
  #pragma unroll
  for (int ot = 0; ot < OCW; ++ot) {
    const int ocb = (ocT0 + ot) * 16 + hf * 8;
    float s[8], bb[8];
    #pragma unroll
    for (int r = 0; r < 8; ++r) { s[r] = Sc[ocb + r]; bb[r] = Bc[ocb + r]; }
    #pragma unroll
    for (int nt = 0; nt < 4; ++nt) {
      size_t o = (size_t)(pix0 + nt * 16 + n2) * COUT + ocb;
      unsigned int lo = 0, hi = 0;
      #pragma unroll
      for (int r = 0; r < 8; ++r) {
        float h = (float)acc[ot][nt][r] * s[r] + bb[r];
        float q = fminf(fmaxf(rintf(h * inv), 0.f), 15.f);
        unsigned int u = (unsigned int)q;
        if (r < 4) lo |= u << (8 * r);
        else       hi |= u << (8 * (r - 4));
      }
      uint2 pk; pk.x = lo; pk.y = hi;
      *(uint2*)(out + o) = pk;
    }
  }
}

__global__ void pool_kernel(const uint8_t* __restrict__ in, uint8_t* __restrict__ out,
                            int Hi, int Wi, int C, int total32) {
  int idx = blockIdx.x * blockDim.x + threadIdx.x;
  if (idx >= total32) return;
  int C4 = C >> 2;
  int c4 = idx % C4;
  int sp = idx / C4;
  int wo = Wi >> 1, ho = Hi >> 1;
  int x = sp % wo;
  int y = (sp / wo) % ho;
  int n = sp / (wo * ho);
  size_t base = ((size_t)(n * Hi + 2 * y) * Wi + 2 * x) * C + c4 * 4;
  u8x4 a = *(const u8x4*)(in + base);
  u8x4 b = *(const u8x4*)(in + base + C);
  u8x4 c = *(const u8x4*)(in + base + (size_t)Wi * C);
  u8x4 d = *(const u8x4*)(in + base + (size_t)Wi * C + C);
  u8x4 mx = __builtin_elementwise_max(__builtin_elementwise_max(a, b),
                                      __builtin_elementwise_max(c, d));
  *(u8x4*)(out + (size_t)idx * 4) = mx;
}

__global__ __launch_bounds__(256) void fc_kernel(const uint8_t* __restrict__ act,
                                                 const float* __restrict__ aa5,
                                                 const float* __restrict__ fcw,
                                                 const float* __restrict__ fcb,
                                                 float* __restrict__ out) {
  int gwave = (blockIdx.x * blockDim.x + threadIdx.x) >> 5;
  int lane  = threadIdx.x & 31;
  if (gwave >= 5120) return;
  int n = gwave / 10, cls = gwave % 10;
  const uint8_t* a = act + (size_t)n * 8192;
  const float* wr  = fcw + (size_t)cls * 8192;
  float sum = 0.f;
  for (int i = lane; i < 8192; i += 32) {
    int sidx = i >> 9;
    int ch   = i & 511;
    sum += (float)a[i] * wr[ch * 16 + sidx];
  }
  #pragma unroll
  for (int off = 16; off > 0; off >>= 1) sum += __shfl_xor(sum, off, 32);
  if (lane == 0) out[n * 10 + cls] = sum * aa5[0] + fcb[cls];
}

extern "C" void kernel_launch(void* const* d_in, const int* in_sizes, int n_in,
                              void* d_out, int out_size, void* d_ws, size_t ws_size,
                              hipStream_t stream) {
  (void)in_sizes; (void)n_in; (void)out_size; (void)ws_size;
  const float* x = (const float*)d_in[0];
  const float *Wt[6], *WA[6], *G[6], *Bp[6], *M[6], *V[6], *AA[6];
  Wt[0] = (const float*)d_in[1]; WA[0] = nullptr;
  G[0]  = (const float*)d_in[2]; Bp[0] = (const float*)d_in[3];
  M[0]  = (const float*)d_in[4]; V[0]  = (const float*)d_in[5];
  AA[0] = (const float*)d_in[6];
  for (int i = 1; i < 6; ++i) {
    int base = 7 + (i - 1) * 7;
    Wt[i] = (const float*)d_in[base + 0];
    WA[i] = (const float*)d_in[base + 1];
    G[i]  = (const float*)d_in[base + 2];
    Bp[i] = (const float*)d_in[base + 3];
    M[i]  = (const float*)d_in[base + 4];
    V[i]  = (const float*)d_in[base + 5];
    AA[i] = (const float*)d_in[base + 6];
  }
  const float* fcw = (const float*)d_in[42];
  const float* fcb = (const float*)d_in[43];

  char* ws = (char*)d_ws;
  uint8_t* ping = (uint8_t*)(ws + PING_OFF);
  uint8_t* pong = (uint8_t*)(ws + PONG_OFF);

  static const int CIN[6]  = {3, 128, 128, 256, 256, 512};
  static const int COUT[6] = {128, 128, 256, 256, 512, 512};
  static const int LOGC[6] = {0, 7, 7, 8, 8, 9};

  int8_t* wf[6];
  size_t off = WF_OFF;
  for (int i = 1; i < 6; ++i) {
    wf[i] = (int8_t*)(ws + off);
    off += (size_t)COUT[i] * CIN[i] * 9;
  }
  float* Sb = (float*)(ws + SB_OFF);
  float* Sarr[6]; float* Barr[6];
  for (int i = 0; i < 6; ++i) { Sarr[i] = Sb + i * 512; Barr[i] = Sb + 6 * 512 + i * 512; }

  for (int i = 0; i < 6; ++i) {
    int co = COUT[i];
    scalebias_kernel<<<(co + 255) / 256, 256, 0, stream>>>(
        G[i], Bp[i], M[i], V[i], WA[i], i ? AA[i - 1] : nullptr,
        Sarr[i], Barr[i], co);
  }
  for (int i = 1; i < 6; ++i) {
    int total = COUT[i] * CIN[i] * 9;
    wquant_kernel<<<(total + 255) / 256, 256, 0, stream>>>(
        Wt[i], WA[i], wf[i], CIN[i], LOGC[i], total);
  }
  conv0_kernel<<<262144, 256, 0, stream>>>(x, Wt[0], Sarr[0], Barr[0], AA[0], ping);
  convq_kernel<128, 128, 32, 32, 1><<<dim3(8192, 1), 256, 0, stream>>>(
      ping, wf[1], Sarr[1], Barr[1], AA[1], pong);
  pool_kernel<<<16384, 256, 0, stream>>>(pong, ping, 32, 32, 128, 512 * 16 * 16 * 32);
  convq_kernel<128, 256, 16, 16, 2><<<dim3(2048, 1), 256, 0, stream>>>(
      ping, wf[2], Sarr[2], Barr[2], AA[2], pong);
  convq_kernel<256, 256, 16, 16, 2><<<dim3(2048, 1), 256, 0, stream>>>(
      pong, wf[3], Sarr[3], Barr[3], AA[3], ping);
  pool_kernel<<<8192, 256, 0, stream>>>(ping, pong, 16, 16, 256, 512 * 8 * 8 * 64);
  convq_kernel<256, 512, 8, 8, 2><<<dim3(512, 2), 256, 0, stream>>>(
      pong, wf[4], Sarr[4], Barr[4], AA[4], ping);
  convq_kernel<512, 512, 8, 8, 2><<<dim3(512, 2), 256, 0, stream>>>(
      ping, wf[5], Sarr[5], Barr[5], AA[5], pong);
  pool_kernel<<<4096, 256, 0, stream>>>(pong, ping, 8, 8, 512, 512 * 4 * 4 * 128);
  fc_kernel<<<640, 256, 0, stream>>>(ping, AA[5], fcw, fcb, (float*)d_out);
}
